// FlashAttention_52201032515805
// MI455X (gfx1250) — matrix operand order repaired and checked by reading, not compiled, not run on hardware
//
#include <hip/hip_runtime.h>
#include <stdint.h>

// ---------------------------------------------------------------------------
// Block-causal MHA for MI455X (gfx1250), bf16 WMMA with fp32 accumulation.
// Pipeline: cvt(f32->bf16) -> QKV GEMM -> flash attention -> out-proj GEMM.
// Matrix math on v_wmma_f32_16x16x32_bf16; K tiles staged into LDS with
// global_load_async_to_lds_b128 (ASYNCcnt) in the attention kernel.
// ---------------------------------------------------------------------------

#define BATCH 2
#define SEQ   2048
#define CDIM  1024
#define NH    16
#define HD    64
#define QBLK  64
#define ROWS  (BATCH * SEQ)   // 4096

typedef __bf16 bf16_t;
typedef __attribute__((ext_vector_type(16))) __bf16 v16bf;
typedef __attribute__((ext_vector_type(8)))  __bf16 v8bf;
typedef __attribute__((ext_vector_type(8)))  float  v8f;

union Frag16 { v16bf v; v8bf h[2]; };

// A-matrix fragment (16x32 bf16): lane holds row m = lane&15,
// K = 16 contiguous values starting at (lane>=16 ? 16 : 0). p points at K start.
__device__ __forceinline__ v16bf frag_a(const bf16_t* p) {
    Frag16 f;
    f.h[0] = *(const v8bf*)(p);
    f.h[1] = *(const v8bf*)(p + 8);
    return f.v;
}
// B-matrix fragment (32x16 bf16): lane holds column n = lane&15,
// K = 16 contiguous values starting at (lane>=16 ? 16 : 0). p points at K start.
__device__ __forceinline__ v16bf frag_b(const bf16_t* p) {
    Frag16 f;
    f.h[0] = *(const v8bf*)(p);
    f.h[1] = *(const v8bf*)(p + 8);
    return f.v;
}

__device__ __forceinline__ v8f wmma_bf16(v16bf a, v16bf b, v8f c) {
    return __builtin_amdgcn_wmma_f32_16x16x32_bf16(
        /*neg_a=*/false, a, /*neg_b=*/false, b,
        /*c_mod=*/(short)0, c, /*reuse_a=*/false, /*reuse_b=*/false);
}

// Async DMA: 16 bytes global -> LDS, tracked by ASYNCcnt.
__device__ __forceinline__ void async_g2l_b128(uint32_t lds_byte_off, const void* gaddr) {
    asm volatile("global_load_async_to_lds_b128 %0, %1, off"
                 :: "v"(lds_byte_off), "v"(gaddr) : "memory");
}
__device__ __forceinline__ void wait_asynccnt0() {
    asm volatile("s_wait_asynccnt 0x0" ::: "memory");
}

// ---------------------------------------------------------------------------
__global__ void cvt_f32_bf16(const float* __restrict__ s, bf16_t* __restrict__ d, int n) {
    int i      = blockIdx.x * blockDim.x + threadIdx.x;
    int stride = gridDim.x * blockDim.x;
    for (; i < n; i += stride) d[i] = (bf16_t)s[i];
}

// ---------------------------------------------------------------------------
// QKV projection: [4096 x 1024] x [3072 x 1024]^T, scatter into q/k/v [B,H,N,hd].
// 128 threads = 4 waves; each wave 32x64 output (2x4 WMMA tiles); WG tile 64x128.
__global__ __launch_bounds__(128)
void qkv_gemm(const bf16_t* __restrict__ xb, const bf16_t* __restrict__ wb,
              bf16_t* __restrict__ q, bf16_t* __restrict__ k, bf16_t* __restrict__ v) {
    const int lane   = threadIdx.x & 31;
    const int wv     = threadIdx.x >> 5;
    const int lanelo = lane & 15;
    const int half   = lane >> 4;
    const int row0   = blockIdx.y * 64  + (wv >> 1) * 32;
    const int col0   = blockIdx.x * 128 + (wv & 1) * 64;

    v8f acc[2][4];
    v8f zero = {};
    #pragma unroll
    for (int mt = 0; mt < 2; ++mt)
        #pragma unroll
        for (int nt = 0; nt < 4; ++nt) acc[mt][nt] = zero;

    for (int k0 = 0; k0 < CDIM; k0 += 32) {
        v16bf af[2], bfr[4];
        #pragma unroll
        for (int mt = 0; mt < 2; ++mt)
            af[mt] = frag_a(xb + (size_t)(row0 + mt * 16 + lanelo) * CDIM + k0 + half * 16);
        #pragma unroll
        for (int nt = 0; nt < 4; ++nt)
            bfr[nt] = frag_b(wb + (size_t)(col0 + nt * 16 + lanelo) * CDIM + k0 + half * 16);
        #pragma unroll
        for (int mt = 0; mt < 2; ++mt)
            #pragma unroll
            for (int nt = 0; nt < 4; ++nt)
                acc[mt][nt] = wmma_bf16(af[mt], bfr[nt], acc[mt][nt]);
    }

    // Scatter: column c -> (s = c>>10, h = (c>>6)&15, d = c&63); q pre-scaled.
    #pragma unroll
    for (int nt = 0; nt < 4; ++nt) {
        const int c = col0 + nt * 16 + lanelo;
        const int s = c >> 10;
        const int h = (c >> 6) & (NH - 1);
        const int d = c & (HD - 1);
        bf16_t* dst = (s == 0) ? q : (s == 1) ? k : v;
        const float scl = (s == 0) ? 0.125f : 1.0f;   // hd^-0.5 = 1/8
        #pragma unroll
        for (int mt = 0; mt < 2; ++mt) {
            #pragma unroll
            for (int r = 0; r < 8; ++r) {
                const int row = row0 + mt * 16 + half * 8 + r;
                const int b   = row >> 11;           // /SEQ
                const int n   = row & (SEQ - 1);
                dst[(((size_t)(b * NH + h) * SEQ) + n) * HD + d] =
                    (bf16_t)(acc[mt][nt][r] * scl);
            }
        }
    }
}

// ---------------------------------------------------------------------------
// Flash attention, block-causal (query block qb attends key blocks 0..qb fully).
// One WG per (b, h, qb); 4 waves, each owns 16 query rows x full hd=64.
// K tile staged once per block via async DMA to LDS; V staged transposed.
__global__ __launch_bounds__(128)
void attn_kernel(const bf16_t* __restrict__ q, const bf16_t* __restrict__ k,
                 const bf16_t* __restrict__ v, bf16_t* __restrict__ attn) {
    const int qb = blockIdx.x, h = blockIdx.y, b = blockIdx.z;
    const int lane   = threadIdx.x & 31;
    const int wv     = threadIdx.x >> 5;
    const int lanelo = lane & 15;
    const int half   = lane >> 4;

    __shared__ __align__(16) bf16_t KT[QBLK * HD];     // K tile row-major [key][dim]
    __shared__ __align__(16) bf16_t VT[HD * QBLK];     // V transposed: [dim][key]
    __shared__ __align__(16) bf16_t P[4][16 * QBLK];   // per-wave P tile [row][key]

    const size_t headbase = (size_t)(b * NH + h) * SEQ * HD;
    const int qr0 = qb * QBLK + wv * 16;
    const uint32_t kt_lds0 = (uint32_t)(uintptr_t)&KT[0];  // LDS byte offset (aperture low bits)

    // Q fragments (pre-scaled), hd=64 -> two K-chunks of 32.
    v16bf aq[2];
    #pragma unroll
    for (int c = 0; c < 2; ++c)
        aq[c] = frag_a(q + headbase + (size_t)(qr0 + lanelo) * HD + c * 32 + half * 16);

    v8f o[4];
    v8f zero = {};
    float mrow[8], lrow[8];
    #pragma unroll
    for (int t = 0; t < 4; ++t) o[t] = zero;
    #pragma unroll
    for (int r = 0; r < 8; ++r) { mrow[r] = -__builtin_inff(); lrow[r] = 0.f; }

    for (int j = 0; j <= qb; ++j) {
        __syncthreads();   // previous iteration's KT/VT/P reads complete

        // Issue async DMA of the 8KB K tile (contiguous rows) into LDS.
        {
            const char* gk = (const char*)(k + headbase + (size_t)j * QBLK * HD);
            #pragma unroll
            for (int p4 = 0; p4 < 4; ++p4) {
                const uint32_t boff = (uint32_t)(p4 * 2048 + threadIdx.x * 16);
                async_g2l_b128(kt_lds0 + boff, gk + boff);
            }
        }

        // Stage V^T cooperatively (overlaps the async K DMA).
        for (int idx = threadIdx.x; idx < QBLK * HD; idx += 128) {
            const int key = idx >> 6, d = idx & (HD - 1);
            VT[d * QBLK + key] = v[headbase + (size_t)(j * QBLK + key) * HD + d];
        }

        wait_asynccnt0();   // this wave's K DMA complete
        __syncthreads();    // KT + VT visible to all waves

        // S = Q * K^T : K rows are feature-contiguous -> B-frags from LDS.
        v8f s[4];
        #pragma unroll
        for (int kt = 0; kt < 4; ++kt) s[kt] = zero;
        #pragma unroll
        for (int c = 0; c < 2; ++c) {
            #pragma unroll
            for (int kt = 0; kt < 4; ++kt) {
                v16bf bk = frag_b(&KT[(kt * 16 + lanelo) * HD + c * 32 + half * 16]);
                s[kt] = wmma_bf16(aq[c], bk, s[kt]);
            }
        }

        // Online softmax. Rows live in VGPR index r (group row = half*8 + r),
        // columns in lanes: xor-shuffles 1/2/4/8 reduce within each 16-lane half.
        #pragma unroll
        for (int r = 0; r < 8; ++r) {
            float rm = s[0][r];
            #pragma unroll
            for (int kt = 1; kt < 4; ++kt) rm = fmaxf(rm, s[kt][r]);
            #pragma unroll
            for (int m = 1; m < 16; m <<= 1) rm = fmaxf(rm, __shfl_xor(rm, m, 32));
            const float nm    = fmaxf(mrow[r], rm);
            const float alpha = __expf(mrow[r] - nm);
            mrow[r] = nm;
            float rs = 0.f;
            #pragma unroll
            for (int kt = 0; kt < 4; ++kt) {
                const float e = __expf(s[kt][r] - nm);
                s[kt][r] = e;
                rs += e;
            }
            #pragma unroll
            for (int m = 1; m < 16; m <<= 1) rs += __shfl_xor(rs, m, 32);
            lrow[r] = lrow[r] * alpha + rs;
            #pragma unroll
            for (int t = 0; t < 4; ++t) o[t][r] *= alpha;
        }

        // Spill P (D layout) to LDS so it can be re-read in A layout.
        #pragma unroll
        for (int kt = 0; kt < 4; ++kt)
            #pragma unroll
            for (int r = 0; r < 8; ++r)
                P[wv][(half * 8 + r) * QBLK + kt * 16 + lanelo] = (bf16_t)s[kt][r];

        __syncthreads();   // P visible

        // O += P * V  (keys = K dim, 2 chunks of 32).
        #pragma unroll
        for (int c = 0; c < 2; ++c) {
            v16bf ap = frag_a(&P[wv][lanelo * QBLK + c * 32 + half * 16]);
            #pragma unroll
            for (int t = 0; t < 4; ++t) {
                v16bf bv = frag_b(&VT[(t * 16 + lanelo) * QBLK + c * 32 + half * 16]);
                o[t] = wmma_bf16(ap, bv, o[t]);
            }
        }

        // Prefetch next V block (gfx1250 global_prefetch_b8 path).
        if (j < qb)
            __builtin_prefetch(v + headbase + (size_t)((j + 1) * QBLK + lane) * HD, 0, 1);
    }

    // Epilogue: normalize and write attn [B,N,C] bf16.
    #pragma unroll
    for (int t = 0; t < 4; ++t) {
        const int col = h * HD + t * 16 + lanelo;
        #pragma unroll
        for (int r = 0; r < 8; ++r) {
            const int n = qr0 + half * 8 + r;
            attn[((size_t)b * SEQ + n) * CDIM + col] = (bf16_t)(o[t][r] / lrow[r]);
        }
    }
}

// ---------------------------------------------------------------------------
// Output projection: out = attn @ w_proj^T + b_proj, fp32 result.
// Same 32x64-per-wave tiling as qkv_gemm.
__global__ __launch_bounds__(128)
void proj_gemm(const bf16_t* __restrict__ attn, const bf16_t* __restrict__ wb,
               const float* __restrict__ bias, float* __restrict__ out) {
    const int lane   = threadIdx.x & 31;
    const int wv     = threadIdx.x >> 5;
    const int lanelo = lane & 15;
    const int half   = lane >> 4;
    const int row0   = blockIdx.y * 64  + (wv >> 1) * 32;
    const int col0   = blockIdx.x * 128 + (wv & 1) * 64;

    v8f acc[2][4];
    v8f zero = {};
    #pragma unroll
    for (int mt = 0; mt < 2; ++mt)
        #pragma unroll
        for (int nt = 0; nt < 4; ++nt) acc[mt][nt] = zero;

    for (int k0 = 0; k0 < CDIM; k0 += 32) {
        v16bf af[2], bfr[4];
        #pragma unroll
        for (int mt = 0; mt < 2; ++mt)
            af[mt] = frag_a(attn + (size_t)(row0 + mt * 16 + lanelo) * CDIM + k0 + half * 16);
        #pragma unroll
        for (int nt = 0; nt < 4; ++nt)
            bfr[nt] = frag_b(wb + (size_t)(col0 + nt * 16 + lanelo) * CDIM + k0 + half * 16);
        #pragma unroll
        for (int mt = 0; mt < 2; ++mt)
            #pragma unroll
            for (int nt = 0; nt < 4; ++nt)
                acc[mt][nt] = wmma_bf16(af[mt], bfr[nt], acc[mt][nt]);
    }

    #pragma unroll
    for (int nt = 0; nt < 4; ++nt) {
        const int   col = col0 + nt * 16 + lanelo;
        const float bc  = bias[col];
        #pragma unroll
        for (int mt = 0; mt < 2; ++mt)
            #pragma unroll
            for (int r = 0; r < 8; ++r) {
                const int row = row0 + mt * 16 + half * 8 + r;
                out[(size_t)row * CDIM + col] = acc[mt][nt][r] + bc;
            }
    }
}

// ---------------------------------------------------------------------------
extern "C" void kernel_launch(void* const* d_in, const int* in_sizes, int n_in,
                              void* d_out, int out_size, void* d_ws, size_t ws_size,
                              hipStream_t stream) {
    (void)in_sizes; (void)n_in; (void)out_size; (void)ws_size;
    const float* x      = (const float*)d_in[0];   // [2,2048,1024]
    const float* w_qkv  = (const float*)d_in[1];   // [3072,1024]
    const float* w_proj = (const float*)d_in[2];   // [1024,1024]
    const float* b_proj = (const float*)d_in[3];   // [1024]
    float*       out    = (float*)d_out;           // [2,2048,1024] fp32

    // Workspace layout (~40 MB; attn aliases consumed xb region):
    char*   ws     = (char*)d_ws;
    bf16_t* xb     = (bf16_t*)(ws + (size_t)0);          // 8 MB  (x bf16)
    bf16_t* wqkvb  = (bf16_t*)(ws + ((size_t)8  << 20)); // 6 MB
    bf16_t* wprojb = (bf16_t*)(ws + ((size_t)14 << 20)); // 2 MB
    bf16_t* qd     = (bf16_t*)(ws + ((size_t)16 << 20)); // 8 MB [B,H,N,hd]
    bf16_t* kd     = (bf16_t*)(ws + ((size_t)24 << 20)); // 8 MB
    bf16_t* vd     = (bf16_t*)(ws + ((size_t)32 << 20)); // 8 MB
    bf16_t* attn   = xb;                                  // reuse after qkv_gemm

    cvt_f32_bf16<<<1024, 256, 0, stream>>>(x,      xb,     ROWS * CDIM);
    cvt_f32_bf16<<<1024, 256, 0, stream>>>(w_qkv,  wqkvb,  3 * CDIM * CDIM);
    cvt_f32_bf16<<<512,  256, 0, stream>>>(w_proj, wprojb, CDIM * CDIM);

    qkv_gemm<<<dim3(3 * CDIM / 128, ROWS / 64), 128, 0, stream>>>(xb, wqkvb, qd, kd, vd);
    attn_kernel<<<dim3(SEQ / QBLK, NH, BATCH), 128, 0, stream>>>(qd, kd, vd, attn);
    proj_gemm<<<dim3(CDIM / 128, ROWS / 64), 128, 0, stream>>>(attn, wprojb, b_proj, out);
}
